// LSTMCell_37924561224179
// MI455X (gfx1250) — matrix operand order repaired and checked by reading, not compiled, not run on hardware
//
#include <hip/hip_runtime.h>
#include <hip/hip_bf16.h>
#include <math.h>

// Problem dims
#define BDIM 8192
#define DDIM 4096   // K
#define HDIM 2048

// GEMM tiling
#define BM 128
#define BK 64
#define LDA 72      // BK + 8 bf16 padding (144B row stride -> conflict-spread)
#define LDG 132     // gates LDS row stride in floats
#define NBLK (DDIM / BK)

#define ABUF_BYTES (BM * LDA * 2)          // 18432
#define STAGE_BYTES (2 * ABUF_BYTES)       // A+B per buffer = 36864

typedef __attribute__((ext_vector_type(16))) __bf16 v16bf;
typedef __attribute__((ext_vector_type(8)))  __bf16 v8bf;
typedef __attribute__((ext_vector_type(8)))  float  v8f;
typedef int v4i_ __attribute__((vector_size(16)));

union FragAB { v16bf v; v8bf h[2]; };

#if defined(__has_builtin)
#if __has_builtin(__builtin_amdgcn_global_load_async_to_lds_b128)
#define HAVE_ASYNC_LDS 1
#endif
#endif

#if defined(__has_builtin) && __has_builtin(__builtin_amdgcn_s_wait_asynccnt)
#define WAIT_ASYNC0() __builtin_amdgcn_s_wait_asynccnt(0)
#else
#define WAIT_ASYNC0() asm volatile("s_wait_asynccnt 0x0" ::: "memory")
#endif

#ifdef HAVE_ASYNC_LDS
// builtin signature (from clang diagnostic): param0 = AS1 int4* (non-const), param1 = LDS ptr
#define GLBP(p) ((__attribute__((address_space(1))) v4i_*)(p))
#define LDSP(p) ((__attribute__((address_space(3))) v4i_*)(p))
#endif

__device__ __forceinline__ float sigmoidf_(float x) {
    return 1.0f / (1.0f + __expf(-x));
}

// -------- Phase 1a: combine = bf16(input + hidden), [B, D] --------
__global__ __launch_bounds__(256) void combine_cast_kernel(
    const float* __restrict__ in, const float* __restrict__ hid,
    __bf16* __restrict__ cbf, int n4)
{
    int i = blockIdx.x * blockDim.x + threadIdx.x;
    if (i >= n4) return;
    float4 a = ((const float4*)in)[i];
    float4 b = ((const float4*)hid)[i];
    union { __bf16 e[4]; uint2 u; } o;
    o.e[0] = (__bf16)(a.x + b.x);
    o.e[1] = (__bf16)(a.y + b.y);
    o.e[2] = (__bf16)(a.z + b.z);
    o.e[3] = (__bf16)(a.w + b.w);
    ((uint2*)cbf)[i] = o.u;
}

// -------- Phase 1b: Wt[nOff + h][k] = bf16(Wg[k][h])  (N-major weights) --------
__global__ __launch_bounds__(256) void wtrans_cast_kernel(
    const float* __restrict__ Wg, __bf16* __restrict__ Wt, int nOff)
{
    __shared__ __bf16 tile[32][33];
    int k0 = blockIdx.x * 32, h0 = blockIdx.y * 32;
    int tx = threadIdx.x, ty = threadIdx.y;   // 32 x 8
#pragma unroll
    for (int q = 0; q < 4; ++q)
        tile[ty + 8 * q][tx] = (__bf16)Wg[(size_t)(k0 + ty + 8 * q) * HDIM + h0 + tx];
    __syncthreads();
#pragma unroll
    for (int q = 0; q < 4; ++q)
        Wt[(size_t)(nOff + h0 + ty + 8 * q) * DDIM + k0 + tx] = tile[tx][ty + 8 * q];
}

// 16 x v_wmma_f32_16x16x32_bf16 over one BK=64 tile pair in LDS
#define COMPUTE_TILE(A_, B_)                                                     \
    {                                                                            \
        _Pragma("unroll")                                                        \
        for (int ks = 0; ks < BK; ks += 32) {                                    \
            FragAB bfrag[2];                                                     \
            _Pragma("unroll")                                                    \
            for (int s = 0; s < 2; ++s) {                                        \
                int c  = wn * 32 + s * 16 + (lane & 15);                         \
                int kb = (lane >> 4) * 16;                                       \
                bfrag[s].h[0] = *(const v8bf*)&(B_)[c * LDA + ks + kb];          \
                bfrag[s].h[1] = *(const v8bf*)&(B_)[c * LDA + ks + kb + 8];      \
            }                                                                    \
            _Pragma("unroll")                                                    \
            for (int t = 0; t < 4; ++t) {                                        \
                FragAB afrag;                                                    \
                int r  = wm * 64 + t * 16 + (lane & 15);                         \
                int ka = (lane >> 4) * 16;                                       \
                afrag.h[0] = *(const v8bf*)&(A_)[r * LDA + ks + ka];             \
                afrag.h[1] = *(const v8bf*)&(A_)[r * LDA + ks + ka + 8];         \
                _Pragma("unroll")                                                \
                for (int s = 0; s < 2; ++s) {                                    \
                    acc[t][s] = __builtin_amdgcn_wmma_f32_16x16x32_bf16(         \
                        false, afrag.v, false, bfrag[s].v,                       \
                        (short)0, acc[t][s], false, false);                      \
                }                                                                \
            }                                                                    \
        }                                                                        \
    }

// -------- Phase 2: fused bf16 WMMA GEMM + LSTM epilogue --------
// grid = (B/BM, H/32); block = 256 (8 waves, 2x4 wave grid, 64x32 per wave)
// N tile (128 cols) = 4 gates x 32 h-columns: n_local = g*32 + j -> n_global = g*2048 + h0 + j
__global__ __launch_bounds__(256) void lstm_gemm_fused_kernel(
    const __bf16* __restrict__ Acb,   // [B, D]   bf16 combine
    const __bf16* __restrict__ Wt,    // [4H, D]  bf16 N-major weights
    const float*  __restrict__ cell,  // [B, H]
    const float*  __restrict__ bfv, const float* __restrict__ bcv,
    const float*  __restrict__ biv, const float* __restrict__ bov,
    float* __restrict__ outp, float* __restrict__ hsp, float* __restrict__ csp)
{
    // staging: 2 x (A tile 18432B + B tile 18432B) = 73728B; gates reuse = 67584B
    __shared__ __attribute__((aligned(16))) char smem[2 * STAGE_BYTES];

    const int tid  = threadIdx.x;
    const int lane = tid & 31;
    const int wid  = tid >> 5;
    const int wm   = wid & 1;    // wave row (0..1) -> 64 M rows
    const int wn   = wid >> 1;   // wave col (0..3) -> 32 N cols
    const int m0   = blockIdx.x * BM;
    const int h0   = blockIdx.y * 32;

    // per-thread 16B staging chunk, q-th chunk = +32 rows (row<32, so B chunk q is gate q)
    const int row   = tid >> 3;           // 0..31
    const int kc    = (tid & 7) * 8;      // 0..56 (bf16 elems)
    const int sOffB = (row * LDA + kc) * 2;               // LDS byte offset of chunk 0
    const __bf16* aB = Acb + (size_t)(m0 + row) * DDIM + kc;   // + q*32*DDIM
    const __bf16* bB = Wt  + (size_t)(h0 + row) * DDIM + kc;   // + q*HDIM*DDIM

    v8f acc[4][2] = {};

#ifdef HAVE_ASYNC_LDS
    // ---- async global->LDS staging (ASYNCcnt), double buffered ----
#define STAGE_TILE_ASYNC(bufbase, koff)                                              \
    {                                                                                \
        char* la = (bufbase);                                                        \
        char* lb = (bufbase) + ABUF_BYTES;                                           \
        _Pragma("unroll")                                                            \
        for (int q = 0; q < 4; ++q) {                                                \
            __builtin_amdgcn_global_load_async_to_lds_b128(                          \
                GLBP(aB + (size_t)q * 32 * DDIM + (koff)),                           \
                LDSP(la + sOffB + q * 32 * LDA * 2), 0, 0);                          \
            __builtin_amdgcn_global_load_async_to_lds_b128(                          \
                GLBP(bB + (size_t)q * HDIM * DDIM + (koff)),                         \
                LDSP(lb + sOffB + q * 32 * LDA * 2), 0, 0);                          \
        }                                                                            \
    }

    STAGE_TILE_ASYNC(smem, 0);
    WAIT_ASYNC0();
    __syncthreads();

    int buf = 0;
    for (int blk = 0; blk < NBLK; ++blk) {
        if (blk + 1 < NBLK)
            STAGE_TILE_ASYNC(smem + (buf ^ 1) * STAGE_BYTES, (blk + 1) * BK);
        if (blk + 2 < NBLK) {
            __builtin_prefetch(aB + (size_t)(blk + 2) * BK, 0, 0);
            __builtin_prefetch(bB + (size_t)(blk + 2) * BK, 0, 0);
        }

        const __bf16* A_ = (const __bf16*)(smem + buf * STAGE_BYTES);
        const __bf16* B_ = (const __bf16*)(smem + buf * STAGE_BYTES + ABUF_BYTES);
        COMPUTE_TILE(A_, B_);

        if (blk + 1 < NBLK) {
            WAIT_ASYNC0();
            __syncthreads();
            buf ^= 1;
        }
    }
#else
#warning "global_load_async_to_lds_b128 builtin unavailable; using sync single-buffer staging"
    for (int blk = 0; blk < NBLK; ++blk) {
        uint4 va[4], vb[4];
#pragma unroll
        for (int q = 0; q < 4; ++q) {
            va[q] = *(const uint4*)(aB + (size_t)q * 32 * DDIM + blk * BK);
            vb[q] = *(const uint4*)(bB + (size_t)q * HDIM * DDIM + blk * BK);
        }
        __syncthreads();
        {
            char* la = smem;
            char* lb = smem + ABUF_BYTES;
#pragma unroll
            for (int q = 0; q < 4; ++q) {
                *(uint4*)(la + sOffB + q * 32 * LDA * 2) = va[q];
                *(uint4*)(lb + sOffB + q * 32 * LDA * 2) = vb[q];
            }
        }
        __syncthreads();
        const __bf16* A_ = (const __bf16*)(smem);
        const __bf16* B_ = (const __bf16*)(smem + ABUF_BYTES);
        COMPUTE_TILE(A_, B_);
    }
#endif

    // ---- epilogue: dump accumulators into LDS gate buffer (reuses staging) ----
    __syncthreads();
    float* Gs = (float*)smem;    // [128][LDG]
#pragma unroll
    for (int t = 0; t < 4; ++t) {
#pragma unroll
        for (int s = 0; s < 2; ++s) {
            int mrow = wm * 64 + t * 16 + (lane >> 4) * 8;   // C/D layout: VGPR r -> M=r(+8)
            int ncol = wn * 32 + s * 16 + (lane & 15);
#pragma unroll
            for (int r = 0; r < 8; ++r)
                Gs[(mrow + r) * LDG + ncol] = acc[t][s][r];
        }
    }
    __syncthreads();

    // 128 m-rows x 32 h-cols = 4096 outputs; 16 per thread
#pragma unroll 4
    for (int e = 0; e < 16; ++e) {
        int idx = tid + 256 * e;       // 0..4095
        int m = idx >> 5, j = idx & 31;
        int hcol = h0 + j;
        size_t oidx = (size_t)(m0 + m) * HDIM + hcol;

        float gf = Gs[m * LDG +       j] + bfv[hcol];
        float gc = Gs[m * LDG +  32 + j] + bcv[hcol];
        float gi = Gs[m * LDG +  64 + j] + biv[hcol];
        float go = Gs[m * LDG +  96 + j] + bov[hcol];

        float f   = sigmoidf_(gf);
        float cb_ = tanhf(gc);
        float it  = sigmoidf_(sigmoidf_(gi));   // double sigmoid, faithful to ref
        float o   = sigmoidf_(go);
        float c   = cell[oidx] * f + cb_ * it;
        float h   = o * tanhf(c);

        outp[oidx] = o;
        hsp[oidx]  = h;
        csp[oidx]  = c;
    }
}

extern "C" void kernel_launch(void* const* d_in, const int* in_sizes, int n_in,
                              void* d_out, int out_size, void* d_ws, size_t ws_size,
                              hipStream_t stream) {
    const float* input  = (const float*)d_in[0];
    const float* hidden = (const float*)d_in[1];
    const float* cell   = (const float*)d_in[2];
    const float* Wf = (const float*)d_in[3];
    const float* bf = (const float*)d_in[4];
    const float* Wc = (const float*)d_in[5];
    const float* bc = (const float*)d_in[6];
    const float* Wi = (const float*)d_in[7];
    const float* bi = (const float*)d_in[8];
    const float* Wo = (const float*)d_in[9];
    const float* bo = (const float*)d_in[10];

    float* outp = (float*)d_out;
    float* hsp  = outp + (size_t)BDIM * HDIM;
    float* csp  = hsp  + (size_t)BDIM * HDIM;

    __bf16* cbf = (__bf16*)d_ws;                    // [B, D]  bf16 : 64 MB
    __bf16* Wt  = cbf + (size_t)BDIM * DDIM;        // [4H, D] bf16 : 64 MB

    int n4 = BDIM * DDIM / 4;
    combine_cast_kernel<<<(n4 + 255) / 256, 256, 0, stream>>>(input, hidden, cbf, n4);

    dim3 tb(32, 8), tg(DDIM / 32, HDIM / 32);
    wtrans_cast_kernel<<<tg, tb, 0, stream>>>(Wf, Wt, 0 * HDIM);
    wtrans_cast_kernel<<<tg, tb, 0, stream>>>(Wc, Wt, 1 * HDIM);
    wtrans_cast_kernel<<<tg, tb, 0, stream>>>(Wi, Wt, 2 * HDIM);
    wtrans_cast_kernel<<<tg, tb, 0, stream>>>(Wo, Wt, 3 * HDIM);

    dim3 gg(BDIM / BM, HDIM / 32);
    lstm_gemm_fused_kernel<<<gg, 256, 0, stream>>>(cbf, Wt, cell,
                                                   bf, bc, bi, bo,
                                                   outp, hsp, csp);
}
